// SparseMultiheadAttention_56367150793341
// MI455X (gfx1250) — matrix operand order repaired and checked by reading, not compiled, not run on hardware
//
#include <hip/hip_runtime.h>
#include <hip/hip_bf16.h>

// ---------------------------------------------------------------------------
// SparseMultiheadAttention on MI455X (gfx1250), wave32 + WMMA bf16
// + async global->LDS double-buffered K/V tile staging.
//   E=256, H=8, hd=32, T=S=1024, B=16
// d_out = [ out (1024*16*256 f32) | attn_avg (16*1024*1024 f32) ]
// ---------------------------------------------------------------------------

typedef __attribute__((ext_vector_type(16))) __bf16 v16bf;
typedef __attribute__((ext_vector_type(8)))  float  v8f;

#define TGT  1024
#define SRC  1024
#define BSZ_ 16
#define NH   8
#define HD   32
#define ED   256

#if __has_builtin(__builtin_amdgcn_global_load_async_to_lds_b128) && \
    __has_builtin(__builtin_amdgcn_s_wait_asynccnt)
#define USE_ASYNC_LDS 1
#else
#define USE_ASYNC_LDS 0
#endif

// ---- CDNA5 WMMA fragment index helpers (contiguous k = 16h + e, A and B match) ----------------
// A 16x32 bf16: lane(0-15)=row M; element e -> K = (upper-half?16:0)+e.
__device__ __forceinline__ int a_kidx(int e, int hf) {
  return (hf ? 16 : 0) + e;
}
// B 32x16 bf16: lane(0-15)=col N; element e -> K = (upper-half?16:0)+e.
__device__ __forceinline__ int b_kidx(int e, int hf) {
  return (hf ? 16 : 0) + e;
}

__device__ __forceinline__ v8f wmma_bf16f32(v16bf a, v16bf b, v8f c) {
  return __builtin_amdgcn_wmma_f32_16x16x32_bf16(false, a, false, b,
                                                 (short)0, c, false, false);
}

__device__ __forceinline__ float wave_max32(float v) {
  #pragma unroll
  for (int off = 16; off > 0; off >>= 1) v = fmaxf(v, __shfl_xor(v, off, 32));
  return v;
}
__device__ __forceinline__ float wave_sum32(float v) {
  #pragma unroll
  for (int off = 16; off > 0; off >>= 1) v += __shfl_xor(v, off, 32);
  return v;
}

#if USE_ASYNC_LDS
// builtin signature (from toolchain diagnostic): param0 = int4-vector pointer
// in global (device) AS, param1 = int4-vector pointer in LDS AS, imm offset,
// imm cpol.
typedef int v4i_t __attribute__((vector_size(16)));
typedef __attribute__((address_space(1))) v4i_t gas_v4i;
typedef __attribute__((address_space(3))) v4i_t las_v4i;

// one b128 per lane: global -> LDS (async engine, tracked by ASYNCcnt)
__device__ __forceinline__ void async_b128(const void* g, void* l) {
  __builtin_amdgcn_global_load_async_to_lds_b128(
      (gas_v4i*)(uintptr_t)g,
      (las_v4i*)(unsigned)(uintptr_t)l,
      0, 0);
}
// stage a 16-row x 64-byte tile (rows strided by row_stride bytes in memory)
// into contiguous row-major LDS (row*64).  2 async b128 issues per lane.
__device__ __forceinline__ void stage_tile_16x64(void* dst_lds, const void* src,
                                                 int row_stride, int lane) {
  int r0  = lane >> 2;
  int off = (lane & 3) * 16;
  const char* s = (const char*)src;
  char*       d = (char*)dst_lds;
  async_b128(s + (size_t)r0 * row_stride + off,       d + r0 * 64 + off);
  async_b128(s + (size_t)(r0 + 8) * row_stride + off, d + (r0 + 8) * 64 + off);
}
#endif

// read a 32x16 bf16 B-fragment from a staged row-major 16x(32 bf16) tile
__device__ __forceinline__ v16bf frag_from_tile16x64(const void* tile,
                                                     int lane15, int hf) {
  const __bf16* tb = (const __bf16*)tile;
  v16bf b;
  #pragma unroll
  for (int e = 0; e < 16; ++e) b[e] = tb[lane15 * 32 + (hf ? 16 : 0) + e];
  return b;
}

// ---------------------------------------------------------------------------
// Kernel 0: weights fp32 -> bf16
// ---------------------------------------------------------------------------
__global__ __launch_bounds__(256) void cvt_weights_kernel(
    const float* __restrict__ w_in, const float* __restrict__ w_out,
    __bf16* __restrict__ w_in_bf, __bf16* __restrict__ w_out_bf) {
  int i = blockIdx.x * 256 + threadIdx.x;
  if (i < 768 * 256) w_in_bf[i]  = (__bf16)w_in[i];
  if (i < 256 * 256) w_out_bf[i] = (__bf16)w_out[i];
}

// ---------------------------------------------------------------------------
// Kernel 1: fused QKV projection.  y = x @ W^T + b   (x: [16384,256] f32)
// One wave per 16x16 output tile, 8 WMMAs along K=256.
//  q (scaled)  -> [B*H, T, hd] bf16
//  k           -> [B*H, S, hd] bf16
//  v           -> [B*H, hd, S] bf16  (TRANSPOSED for contiguous PV B-frags)
// ---------------------------------------------------------------------------
__global__ __launch_bounds__(128) void proj_qkv_kernel(
    const float* __restrict__ query, const float* __restrict__ key,
    const float* __restrict__ value,
    const __bf16* __restrict__ w_in_bf, const float* __restrict__ in_bias,
    __bf16* __restrict__ q_bf, __bf16* __restrict__ k_bf,
    __bf16* __restrict__ v_t) {
  int gwave  = blockIdx.x * 4 + (threadIdx.x >> 5);
  int lane   = threadIdx.x & 31;
  int hf     = lane >> 4;
  int lane15 = lane & 15;

  int which = gwave / (1024 * 16);          // 0=q,1=k,2=v
  int rem   = gwave - which * (1024 * 16);
  int m0    = (rem >> 4) * 16;
  int n0    = (rem & 15) * 16;

  const float* x = (which == 0) ? query : (which == 1) ? key : value;

  v8f c = {};
  #pragma unroll
  for (int kc = 0; kc < 8; ++kc) {
    int k0 = kc * HD;
    v16bf a, bw;
    #pragma unroll
    for (int e = 0; e < 16; ++e)
      a[e] = (__bf16)x[(size_t)(m0 + lane15) * ED + k0 + a_kidx(e, hf)];
    #pragma unroll
    for (int e = 0; e < 16; ++e)
      bw[e] = w_in_bf[(size_t)(which * 256 + n0 + lane15) * ED + k0 + b_kidx(e, hf)];
    c = wmma_bf16f32(a, bw, c);
  }

  const float scale = (which == 0) ? 0.17677669529663687f : 1.0f;
  #pragma unroll
  for (int r = 0; r < 8; ++r) {
    int m = m0 + r + (hf ? 8 : 0);      // m = t*B + b
    int e = n0 + lane15;
    float val = (c[r] + in_bias[which * 256 + e]) * scale;
    int t = m >> 4, b = m & 15;
    int h = e >> 5, cc = e & 31;
    if (which == 0)
      q_bf[(size_t)((b * NH + h) * TGT + t) * HD + cc] = (__bf16)val;
    else if (which == 1)
      k_bf[(size_t)((b * NH + h) * SRC + t) * HD + cc] = (__bf16)val;
    else  // transposed: [bh][c][s]; r-loop walks b (s fixed) -> strided stores
      v_t[((size_t)(b * NH + h) * HD + cc) * SRC + t] = (__bf16)val;
  }
}

// ---------------------------------------------------------------------------
// Kernel 2: fused attention per (b, 16-row tgt tile).  8 waves / block.
// LDS: 64KB score strip + 64KB head-average + 8KB PV partials
//      + 16KB async staging (8 waves x double-buffered 1KB tiles) = 152KB.
// QK^T and PV both consume double-buffered async global->LDS tile copies.
// Sparsemax: 24-step binary search on tau + exact refinement (no sort).
// ---------------------------------------------------------------------------
__global__ __launch_bounds__(256) void attn_fused_kernel(
    const __bf16* __restrict__ q_bf, const __bf16* __restrict__ k_bf,
    const __bf16* __restrict__ v_t,
    float* __restrict__ o_f32, float* __restrict__ attn_out) {
  __shared__ float s_s[16 * 1024];                     // score / prob strip
  __shared__ float s_avg[16 * 1024];                   // head-avg accumulator
  __shared__ float s_red[8 * 256];                     // PV cross-wave partials
  __shared__ __align__(16) char s_stage[8][2][1024];   // async staging

  int b      = blockIdx.x & 15;
  int t0     = (blockIdx.x >> 4) * 16;
  int tid    = threadIdx.x;
  int wave   = tid >> 5;
  int lane   = tid & 31;
  int hf     = lane >> 4;
  int lane15 = lane & 15;

  for (int i = tid; i < 16 * 1024; i += 256) s_avg[i] = 0.0f;
  __syncthreads();

  for (int h = 0; h < NH; ++h) {
    int bh = b * NH + h;
    const __bf16* qrow = q_bf + (size_t)bh * TGT * HD;
    const __bf16* krow = k_bf + (size_t)bh * SRC * HD;
    const __bf16* vrow = v_t + (size_t)bh * HD * SRC;  // [hd][S]

    // ---- S = Q K^T : wave w does 16x16 tiles j = w, w+8, ..., w+56 -------
    v16bf aq;
    #pragma unroll
    for (int e = 0; e < 16; ++e)
      aq[e] = qrow[(size_t)(t0 + lane15) * HD + a_kidx(e, hf)];

#if USE_ASYNC_LDS
    {
      char* stg = &s_stage[wave][0][0];
      const char* ksrc = (const char*)krow;            // tile j at byte j*1024
      stage_tile_16x64(stg, ksrc + (size_t)wave * 1024, 64, lane);
      for (int i = 0; i < 8; ++i) {
        int j = wave + 8 * i;
        if (i + 1 < 8) {
          stage_tile_16x64(stg + ((i + 1) & 1) * 1024,
                           ksrc + (size_t)(j + 8) * 1024, 64, lane);
          __builtin_amdgcn_s_wait_asynccnt(2);         // front tile landed
        } else {
          __builtin_amdgcn_s_wait_asynccnt(0);
        }
        v16bf bk = frag_from_tile16x64(stg + (i & 1) * 1024, lane15, hf);
        v8f c = {};
        c = wmma_bf16f32(aq, bk, c);
        int s0 = j * 16;
        #pragma unroll
        for (int r = 0; r < 8; ++r)
          s_s[(r + (hf ? 8 : 0)) * 1024 + s0 + lane15] = c[r];
      }
    }
#else
    for (int i = 0; i < 8; ++i) {
      int s0 = (wave + 8 * i) * 16;
      v16bf bk;
      #pragma unroll
      for (int e = 0; e < 16; ++e)
        bk[e] = krow[(size_t)(s0 + lane15) * HD + b_kidx(e, hf)];
      v8f c = {};
      c = wmma_bf16f32(aq, bk, c);
      #pragma unroll
      for (int r = 0; r < 8; ++r)
        s_s[(r + (hf ? 8 : 0)) * 1024 + s0 + lane15] = c[r];
    }
#endif
    __syncthreads();

    // ---- sparsemax on each 1024-length row; 2 rows per wave --------------
    #pragma unroll
    for (int rr = 0; rr < 2; ++rr) {
      int row = wave * 2 + rr;
      float z[32];
      #pragma unroll
      for (int i = 0; i < 32; ++i) z[i] = s_s[row * 1024 + i * 32 + lane];

      float mx = z[0];
      #pragma unroll
      for (int i = 1; i < 32; ++i) mx = fmaxf(mx, z[i]);
      mx = wave_max32(mx);

      // f(tau)=sum(relu(z-tau)) monotone; f(mx-1)>=1, f(mx)=0.
      float lo = mx - 1.0f, hi = mx;
      for (int it = 0; it < 24; ++it) {
        float mid = 0.5f * (lo + hi);
        float s = 0.0f;
        #pragma unroll
        for (int i = 0; i < 32; ++i) s += fmaxf(z[i] - mid, 0.0f);
        s = wave_sum32(s);
        if (s >= 1.0f) lo = mid; else hi = mid;
      }
      float cnt = 0.0f, sum = 0.0f;
      #pragma unroll
      for (int i = 0; i < 32; ++i)
        if (z[i] > lo) { cnt += 1.0f; sum += z[i]; }
      cnt = wave_sum32(cnt);
      sum = wave_sum32(sum);
      float tau = (sum - 1.0f) / fmaxf(cnt, 1.0f);

      #pragma unroll
      for (int i = 0; i < 32; ++i) {
        float p = fmaxf(z[i] - tau, 0.0f);
        s_s[row * 1024 + i * 32 + lane] = p;
        s_avg[row * 1024 + i * 32 + lane] += p * 0.125f;
      }
    }
    __syncthreads();

    // ---- O = P V : wave -> (ntile = w&1, K-quarter = w>>1) ---------------
    {
      int ntile = wave & 1;
      int kq    = wave >> 1;
      v8f c = {};
#if USE_ASYNC_LDS
      char* stg = &s_stage[wave][0][0];
      // chunk kk tile: 16 rows (n) x 32 k bf16, rows strided SRC*2 bytes
      const char* vsrc = (const char*)(vrow + (size_t)(ntile * 16) * SRC);
      stage_tile_16x64(stg, vsrc + (size_t)(kq * 8) * 64, SRC * 2, lane);
      for (int kk = 0; kk < 8; ++kk) {
        int kc0 = (kq * 8 + kk) * HD;
        if (kk + 1 < 8) {
          stage_tile_16x64(stg + ((kk + 1) & 1) * 1024,
                           vsrc + (size_t)(kq * 8 + kk + 1) * 64, SRC * 2, lane);
          __builtin_amdgcn_s_wait_asynccnt(2);
        } else {
          __builtin_amdgcn_s_wait_asynccnt(0);
        }
        v16bf ap, bv;
        #pragma unroll
        for (int e = 0; e < 16; ++e)
          ap[e] = (__bf16)s_s[lane15 * 1024 + kc0 + a_kidx(e, hf)];
        bv = frag_from_tile16x64(stg + (kk & 1) * 1024, lane15, hf);
        c = wmma_bf16f32(ap, bv, c);
      }
#else
      for (int kk = 0; kk < 8; ++kk) {
        int kc0 = (kq * 8 + kk) * HD;
        v16bf ap, bv;
        #pragma unroll
        for (int e = 0; e < 16; ++e)
          ap[e] = (__bf16)s_s[lane15 * 1024 + kc0 + a_kidx(e, hf)];
        #pragma unroll
        for (int e = 0; e < 16; ++e)  // transposed V: contiguous 16 bf16
          bv[e] = vrow[(size_t)(ntile * 16 + lane15) * SRC + kc0 + b_kidx(e, hf)];
        c = wmma_bf16f32(ap, bv, c);
      }
#endif
      #pragma unroll
      for (int r = 0; r < 8; ++r) s_red[wave * 256 + r * 32 + lane] = c[r];
    }
    __syncthreads();

    if (wave < 2) {  // reduce 4 K-quarter partials per n-tile, write O
      int ntile = wave;
      #pragma unroll
      for (int r = 0; r < 8; ++r) {
        float v = s_red[(ntile + 0) * 256 + r * 32 + lane]
                + s_red[(ntile + 2) * 256 + r * 32 + lane]
                + s_red[(ntile + 4) * 256 + r * 32 + lane]
                + s_red[(ntile + 6) * 256 + r * 32 + lane];
        int t = t0 + r + (hf ? 8 : 0);
        int e = h * HD + ntile * 16 + lane15;
        o_f32[(size_t)(t * BSZ_ + b) * ED + e] = v;   // out-proj input layout
      }
    }
    __syncthreads();
  }

  // ---- head-averaged attention [b][t0..t0+15][:] -------------------------
  float* dst = attn_out + (size_t)b * TGT * SRC + (size_t)t0 * SRC;
  for (int i = tid; i < 16 * 1024; i += 256) dst[i] = s_avg[i];
}

// ---------------------------------------------------------------------------
// Kernel 3: output projection  out = O @ Wo^T + bo   ([16384,256])
// ---------------------------------------------------------------------------
__global__ __launch_bounds__(128) void out_proj_kernel(
    const float* __restrict__ o_f32, const __bf16* __restrict__ w_out_bf,
    const float* __restrict__ out_bias, float* __restrict__ out) {
  int gwave  = blockIdx.x * 4 + (threadIdx.x >> 5);
  int lane   = threadIdx.x & 31;
  int hf     = lane >> 4;
  int lane15 = lane & 15;
  int m0     = (gwave >> 4) * 16;
  int n0     = (gwave & 15) * 16;

  v8f c = {};
  #pragma unroll
  for (int kc = 0; kc < 8; ++kc) {
    int k0 = kc * HD;
    v16bf a, bw;
    #pragma unroll
    for (int e = 0; e < 16; ++e)
      a[e] = (__bf16)o_f32[(size_t)(m0 + lane15) * ED + k0 + a_kidx(e, hf)];
    #pragma unroll
    for (int e = 0; e < 16; ++e)
      bw[e] = w_out_bf[(size_t)(n0 + lane15) * ED + k0 + b_kidx(e, hf)];
    c = wmma_bf16f32(a, bw, c);
  }
  #pragma unroll
  for (int r = 0; r < 8; ++r) {
    int m = m0 + r + (hf ? 8 : 0);
    int e = n0 + lane15;
    out[(size_t)m * ED + e] = c[r] + out_bias[e];
  }
}

// ---------------------------------------------------------------------------
extern "C" void kernel_launch(void* const* d_in, const int* in_sizes, int n_in,
                              void* d_out, int out_size, void* d_ws, size_t ws_size,
                              hipStream_t stream) {
  const float* query = (const float*)d_in[0];
  const float* key   = (const float*)d_in[1];
  const float* value = (const float*)d_in[2];
  const float* w_in  = (const float*)d_in[3];
  const float* b_in  = (const float*)d_in[4];
  const float* w_out = (const float*)d_in[5];
  const float* b_out = (const float*)d_in[6];

  char* ws = (char*)d_ws;
  __bf16* w_in_bf  = (__bf16*)(ws);                    // 768*256*2
  __bf16* w_out_bf = (__bf16*)(ws + (768 * 256) * 2);  // 256*256*2
  size_t off = (768 * 256 + 256 * 256) * 2;
  __bf16* q_bf = (__bf16*)(ws + off); off += (size_t)BSZ_ * NH * TGT * HD * 2;
  __bf16* k_bf = (__bf16*)(ws + off); off += (size_t)BSZ_ * NH * SRC * HD * 2;
  __bf16* v_t  = (__bf16*)(ws + off); off += (size_t)BSZ_ * NH * SRC * HD * 2;
  float*  o_f32 = (float*)(ws + off);                  // 16384*256*4

  float* out      = (float*)d_out;                     // [T,B,E]
  float* attn_out = out + (size_t)TGT * BSZ_ * ED;     // [B,T,S]

  cvt_weights_kernel<<<768, 256, 0, stream>>>(w_in, w_out, w_in_bf, w_out_bf);

  proj_qkv_kernel<<<49152 / 4, 128, 0, stream>>>(query, key, value,
                                                 w_in_bf, b_in,
                                                 q_bf, k_bf, v_t);

  attn_fused_kernel<<<BSZ_ * (TGT / 16), 256, 0, stream>>>(q_bf, k_bf, v_t,
                                                           o_f32, attn_out);

  out_proj_kernel<<<16384 / 4, 128, 0, stream>>>(o_f32, w_out_bf, b_out, out);

  (void)in_sizes; (void)n_in; (void)out_size; (void)ws_size;
}
